// PaiConvDG_10050223472788
// MI455X (gfx1250) — matrix operand order repaired and checked by reading, not compiled, not run on hardware
//
#include <hip/hip_runtime.h>
#include <cstdint>

typedef _Float16 h2   __attribute__((ext_vector_type(2)));
typedef _Float16 h8   __attribute__((ext_vector_type(8)));
typedef _Float16 v16h __attribute__((ext_vector_type(16)));
typedef float    v8f  __attribute__((ext_vector_type(8)));
typedef int      v4i  __attribute__((ext_vector_type(4)));

#define B_    16
#define INC   64
#define N_    2048
#define OUTC  64
#define KNB   20
#define KS_   20
#define BN_   (B_*N_)    /* 32768 */
#define C2    128

#if __has_builtin(__builtin_amdgcn_global_load_async_to_lds_b128)
#define HAVE_ASYNC 1
typedef __attribute__((address_space(1))) v4i* gv4p_t;   // global v4i*
typedef __attribute__((address_space(3))) v4i* lv4p_t;   // LDS v4i*
#endif

/* ---------------- workspace layout (bytes) ----------------
   flatT  f32 [BN][64]      @ 0         (8388608)
   W2h    f16 [64][128]     @ 8388608   (16384)
   gSum   f32 [64]          @ 8404992
   gSq    f32 [64]          @ 8405248
   aArr   f32 [64]          @ 8405504
   cArr   f32 [64]          @ 8405760
----------------------------------------------------------- */

__global__ void prep_kernel(const float* __restrict__ feat,
                            const float* __restrict__ convw,
                            float* __restrict__ flatT,
                            _Float16* __restrict__ w2h,
                            float* __restrict__ gS, float* __restrict__ gQ) {
  int i = blockIdx.x * 256 + threadIdx.x;
  if (i < B_ * INC * N_) {
    int p = i & (N_ - 1);
    int c = (i >> 11) & 63;
    int b = i >> 17;
    flatT[(size_t)(b * N_ + p) * INC + c] = feat[i];
  }
  if (i < OUTC * C2) {
    int o = i >> 7, c = i & 127;
    // fold group-shuffle permutation into weights
    w2h[i] = (_Float16)convw[o * C2 + ((c & 31) * 4 + (c >> 5))];
  }
  if (i < OUTC) { gS[i] = 0.f; gQ[i] = 0.f; }
}

__device__ __forceinline__ v16h ldfrag(const _Float16* p0, const _Float16* p1) {
  h8 a = *(const h8*)p0;
  h8 b = *(const h8*)p1;
  return __builtin_shufflevector(a, b, 0,1,2,3,4,5,6,7,8,9,10,11,12,13,14,15);
}

// stage one point's 20x20 f32 permatrix (1600B) into LDS
__device__ __forceinline__ void issueP(const float* __restrict__ src,
                                       float* __restrict__ dst, int lane) {
#ifdef HAVE_ASYNC
  const char* s = (const char*)src;
  char* d = (char*)dst;
  const int off = lane * 16;
  __builtin_amdgcn_global_load_async_to_lds_b128((gv4p_t)(s + off),        (lv4p_t)(d + off),        0, 0);
  __builtin_amdgcn_global_load_async_to_lds_b128((gv4p_t)(s + 512 + off),  (lv4p_t)(d + 512 + off),  0, 0);
  __builtin_amdgcn_global_load_async_to_lds_b128((gv4p_t)(s + 1024 + off), (lv4p_t)(d + 1024 + off), 0, 0);
  if (lane < 4)
    __builtin_amdgcn_global_load_async_to_lds_b128((gv4p_t)(s + 1536 + off), (lv4p_t)(d + 1536 + off), 0, 0);
#else
  for (int k = lane; k < 100; k += 32)
    *(float4*)(dst + 4 * k) = *(const float4*)(src + 4 * k);
#endif
}

#ifdef HAVE_ASYNC
#define WAIT_ASYNC_PEND() asm volatile("s_wait_asynccnt 0x4" ::: "memory")
#define WAIT_ASYNC_ALL()  asm volatile("s_wait_asynccnt 0x0" ::: "memory")
#else
#define WAIT_ASYNC_PEND()
#define WAIT_ASYNC_ALL()
#endif

#define WAVES 4
#define PPW   8

// shared layout: Wl 16384 | accS 256 | accQ 256 | convbL 256 -> header 17152
// per-wave (17536B): Zl 8192 | Pl 2048 | Gl 4096 | Pst 2x1600
#define SMEM_BYTES (17152 + WAVES * 17536)

__global__ __launch_bounds__(128) void paiconv_main(
    const float* __restrict__ flatT, const _Float16* __restrict__ w2h,
    const int*  __restrict__ nb /* int64 viewed as int pairs */,
    const float* __restrict__ perm, const float* __restrict__ convb,
    float* __restrict__ outp, float* __restrict__ gS, float* __restrict__ gQ)
{
  extern __shared__ __align__(16) char smem[];
  _Float16* Wl   = (_Float16*)smem;              // 8192 halves (64x128)
  float* accS    = (float*)(smem + 16384);       // 64
  float* accQ    = (float*)(smem + 16640);       // 64
  float* convbL  = (float*)(smem + 16896);       // 64
  const int tid   = threadIdx.x;
  const int wslot = tid >> 5;
  const int lane  = tid & 31;
  char* wbase  = smem + 17152 + wslot * 17536;
  _Float16* Zl = (_Float16*)wbase;               // 32x128 halves (B of GEMM1)
  _Float16* Pl = (_Float16*)(wbase + 8192);      // 32x32 halves  (B of GEMM2)
  _Float16* Gl = (_Float16*)(wbase + 10240);     // 64x32 halves  (A of GEMM2)
  float*    Pst= (float*)(wbase + 14336);        // 2 x 400 f32 raw P staging

  // cooperative: permuted weights -> LDS, bias -> LDS, zero accumulators
  {
    const uint4* src = (const uint4*)w2h;
    uint4* dst = (uint4*)Wl;
    for (int k = tid; k < 1024; k += 128) dst[k] = src[k];
    if (tid < 64) { accS[tid] = 0.f; accQ[tid] = 0.f; convbL[tid] = convb[tid]; }
  }
  __syncthreads();

  const int lo = lane & 15, hi = lane >> 4;

  // zero padded Z columns (neighbor rows 20..31) once
  {
    uint32_t* z = (uint32_t*)(Zl + 20 * 128);
    for (int k = lane; k < 768; k += 32) z[k] = 0u;
  }

  const int wid = blockIdx.x * WAVES + wslot;
  const v8f vzero = {0.f,0.f,0.f,0.f,0.f,0.f,0.f,0.f};

  // prime double buffer with first point's permatrix
  issueP(perm + (size_t)(wid * PPW) * (KNB * KS_), Pst, lane);

  for (int q = 0; q < PPW; ++q) {
    const int p = wid * PPW + q;

    // issue async fetch of next point's permatrix into the other buffer
    if (q + 1 < PPW)
      issueP(perm + (size_t)(p + 1) * (KNB * KS_), Pst + ((q + 1) & 1) * 400, lane);

    // ---- gather: Z[n][c] = (c<64 ? x0[c] : X[c-64][n]-x0[c-64]) ----
    int myIdx = 0;
    if (lane < KNB) myIdx = nb[((size_t)p * KNB + lane) * 2]; // low dword of int64
    const int idx0 = __shfl(myIdx, 0, 32);
    const float2 x0 = *(const float2*)(flatT + (size_t)idx0 * INC + 2 * lane);
    const h2 x0h = { (_Float16)x0.x, (_Float16)x0.y };
#pragma unroll
    for (int n = 0; n < KNB; ++n) {
      const int idn = __shfl(myIdx, n, 32);
      const float2 xv = *(const float2*)(flatT + (size_t)idn * INC + 2 * lane);
      h2 d = { (_Float16)(xv.x - x0.x), (_Float16)(xv.y - x0.y) };
      *(h2*)(Zl + n * 128 + 2 * lane) = x0h;
      *(h2*)(Zl + n * 128 + 64 + 2 * lane) = d;
    }

    // ---- wait for this point's P tile, transpose to f16 + zero-pad ----
    if (q + 1 < PPW) { WAIT_ASYNC_PEND(); } else { WAIT_ASYNC_ALL(); }
    const float* Ps = Pst + (q & 1) * 400;
#pragma unroll
    for (int kk = 0; kk < 32; kk += 2) {
      float f0 = 0.f, f1 = 0.f;
      if (lane < KS_) {
        if (kk     < KNB) f0 = Ps[kk * KS_ + lane];
        if (kk + 1 < KNB) f1 = Ps[(kk + 1) * KS_ + lane];
      }
      h2 hv = { (_Float16)f0, (_Float16)f1 };
      *(h2*)(Pl + lane * 32 + kk) = hv;
    }
    asm volatile("s_wait_dscnt 0" ::: "memory");

    // ---- GEMM1: G(64x32) = W2(64x128) @ Z(128x32) ----
    v8f acc[4][2];
#pragma unroll
    for (int i = 0; i < 4; ++i) { acc[i][0] = vzero; acc[i][1] = vzero; }
#pragma unroll
    for (int t = 0; t < 4; ++t) {
      // A frags (re-load from LDS each k-step to stay under 256 VGPRs)
      v16h aF[4];
#pragma unroll
      for (int i = 0; i < 4; ++i) {
        const _Float16* pa = Wl + (16 * i + lo) * 128 + 32 * t + hi * 16;
        aF[i] = ldfrag(pa, pa + 8);
      }
      // B frags: lane N = 16j+lo, K = 32t + hi*16 + [0..15]
      const _Float16* pb0 = Zl + lo * 128        + 32 * t + hi * 16;
      const _Float16* pb1 = Zl + (16 + lo) * 128 + 32 * t + hi * 16;
      v16h b0 = ldfrag(pb0, pb0 + 8);
      v16h b1 = ldfrag(pb1, pb1 + 8);
#pragma unroll
      for (int i = 0; i < 4; ++i) {
        acc[i][0] = __builtin_amdgcn_wmma_f32_16x16x32_f16(false, aF[i], false, b0, (short)0, acc[i][0], false, false);
        acc[i][1] = __builtin_amdgcn_wmma_f32_16x16x32_f16(false, aF[i], false, b1, (short)0, acc[i][1], false, false);
      }
    }

    // ---- spill G to LDS as f16 (re-layout D -> A fragments) ----
#pragma unroll
    for (int i = 0; i < 4; ++i)
#pragma unroll
      for (int j = 0; j < 2; ++j)
#pragma unroll
        for (int r = 0; r < 8; ++r) {
          int M = 16 * i + r + (hi ? 8 : 0);
          Gl[M * 32 + 16 * j + lo] = (_Float16)acc[i][j][r];
        }
    asm volatile("s_wait_dscnt 0" ::: "memory");

    // ---- GEMM2: H(64x32) = G(64x32) @ P(32x32) ----
    const _Float16* q0 = Pl + lo * 32        + hi * 16;
    const _Float16* q1 = Pl + (16 + lo) * 32 + hi * 16;
    v16h pb0 = ldfrag(q0, q0 + 8);
    v16h pb1 = ldfrag(q1, q1 + 8);
    v8f hacc[4][2];
#pragma unroll
    for (int i = 0; i < 4; ++i) {
      const _Float16* g0 = Gl + (16 * i + lo) * 32 + hi * 16;
      v16h ga = ldfrag(g0, g0 + 8);
      hacc[i][0] = __builtin_amdgcn_wmma_f32_16x16x32_f16(false, ga, false, pb0, (short)0, vzero, false, false);
      hacc[i][1] = __builtin_amdgcn_wmma_f32_16x16x32_f16(false, ga, false, pb1, (short)0, vzero, false, false);
    }

    // ---- max over s (columns live in lanes), +bias, write, stats ----
    const int bb = p >> 11, pp = p & (N_ - 1);
#pragma unroll
    for (int i = 0; i < 4; ++i)
#pragma unroll
      for (int r = 0; r < 8; ++r) {
        float v = hacc[i][0][r];                                        // s = lo
        float w = (lo < (KS_ - 16)) ? hacc[i][1][r] : -__builtin_inff();// s = 16+lo (valid < 20)
        float m = fmaxf(v, w);
        m = fmaxf(m, __shfl_xor(m, 1, 16));
        m = fmaxf(m, __shfl_xor(m, 2, 16));
        m = fmaxf(m, __shfl_xor(m, 4, 16));
        m = fmaxf(m, __shfl_xor(m, 8, 16));
        int row = 16 * i + r + (hi ? 8 : 0);
        if (lo == 0) {
          float outv = m + convbL[row];
          outp[(size_t)bb * (OUTC * N_) + (size_t)row * N_ + pp] = outv;
          atomicAdd(&accS[row], outv);
          atomicAdd(&accQ[row], outv * outv);
        }
      }
  }

  __syncthreads();
  if (tid < 64) {
    atomicAdd(&gS[tid], accS[tid]);
    atomicAdd(&gQ[tid], accQ[tid]);
  }
}

__global__ void bnstats_kernel(const float* __restrict__ gS, const float* __restrict__ gQ,
                               const float* __restrict__ bnw, const float* __restrict__ bnb,
                               float* __restrict__ aArr, float* __restrict__ cArr) {
  int t = threadIdx.x;
  if (t < OUTC) {
    float mean = gS[t] * (1.f / BN_);
    float var  = gQ[t] * (1.f / BN_) - mean * mean;
    float a = bnw[t] * rsqrtf(var + 1e-5f);
    aArr[t] = a;
    cArr[t] = bnb[t] - mean * a;
  }
}

__global__ void bnapply_kernel(float* __restrict__ out,
                               const float* __restrict__ aArr,
                               const float* __restrict__ cArr) {
  int i = blockIdx.x * 256 + threadIdx.x;
  if (i < B_ * OUTC * N_) {
    int o = (i >> 11) & 63;
    out[i] = out[i] * aArr[o] + cArr[o];
  }
}

extern "C" void kernel_launch(void* const* d_in, const int* in_sizes, int n_in,
                              void* d_out, int out_size, void* d_ws, size_t ws_size,
                              hipStream_t stream) {
  const float* feature = (const float*)d_in[0];
  const int*   neigh   = (const int*)d_in[1];   // int64 entries, read low dword
  const float* perm    = (const float*)d_in[2];
  const float* convw   = (const float*)d_in[3];
  const float* convb   = (const float*)d_in[4];
  const float* bnw     = (const float*)d_in[5];
  const float* bnb     = (const float*)d_in[6];

  char* ws = (char*)d_ws;
  float*    flatT = (float*)(ws + 0);
  _Float16* w2h   = (_Float16*)(ws + 8388608);
  float*    gS    = (float*)(ws + 8404992);
  float*    gQ    = (float*)(ws + 8405248);
  float*    aArr  = (float*)(ws + 8405504);
  float*    cArr  = (float*)(ws + 8405760);

  prep_kernel<<<8192, 256, 0, stream>>>(feature, convw, flatT, w2h, gS, gQ);
  // 1024 blocks x 4 waves x 8 points = 32768 points
  paiconv_main<<<1024, 128, SMEM_BYTES, stream>>>(flatT, w2h, neigh, perm, convb,
                                                  (float*)d_out, gS, gQ);
  bnstats_kernel<<<1, 64, 0, stream>>>(gS, gQ, bnw, bnb, aArr, cArr);
  bnapply_kernel<<<8192, 256, 0, stream>>>((float*)d_out, aArr, cArr);
}
